// HebbianConv2d_81192061764089
// MI455X (gfx1250) — compile-verified, matrix operand order checked by reading, not run on hardware
//
#include <hip/hip_runtime.h>
#include <hip/hip_bf16.h>
#include <cmath>
#include <cstdint>

#define B_   16
#define CIN  64
#define COUT 64
#define H_   128
#define W_   128
#define PADR 2

#define HCOL 132                 // halo columns: 128 + 2*PADR
#define ASTR 72                  // halo row stride in bf16 (144 B, 16B aligned)

typedef __attribute__((ext_vector_type(16))) __bf16 v16bf;
typedef __attribute__((ext_vector_type(8)))  __bf16 v8bf;
typedef __attribute__((ext_vector_type(8)))  float  v8f;
typedef __attribute__((ext_vector_type(4)))  int    v4i;

struct SMK { float v[25]; };

// gfx1250 async global->LDS (signature confirmed by compile probe):
// (v4i AS1*, v4i AS3*, imm offset, imm cpol)
#if __has_builtin(__builtin_amdgcn_global_load_async_to_lds_b128) && \
    __has_builtin(__builtin_amdgcn_s_wait_asynccnt)
#define HAVE_ASYNC_LDS 1
__device__ __forceinline__ void async_copy16(const void* g, void* l) {
    __builtin_amdgcn_global_load_async_to_lds_b128(
        (__attribute__((address_space(1))) v4i*)(void*)g,
        (__attribute__((address_space(3))) v4i*)l, 0, 0);
}
#endif

__device__ __forceinline__ int refl(int i, int n) {
    return i < 0 ? -i : (i >= n ? 2 * n - 2 - i : i);
}

// load a 16-element bf16 fragment: two b128s at +0 and +32 bytes
__device__ __forceinline__ v16bf ld16(const __bf16* p) {
    v8bf lo = *(const v8bf*)p;
    v8bf hi = *(const v8bf*)(p + 16);
    return __builtin_shufflevector(lo, hi, 0,1,2,3,4,5,6,7,8,9,10,11,12,13,14,15);
}

// ---------------------------------------------------------------------------
// Kernel 1: normalized |weight| -> bf16, packed [tap][cout][cin] so a B
// fragment is two contiguous b128 loads per lane.
// ---------------------------------------------------------------------------
__global__ void k_wnorm(const float* __restrict__ w, __bf16* __restrict__ Wb) {
    int tap = blockIdx.x;      // 0..24  (kh*5+kw)
    int o   = threadIdx.x;     // 0..63  (cout)
    float ss = 0.f;
    for (int c = 0; c < CIN; ++c) {
        float v = fabsf(w[(o * CIN + c) * 25 + tap]);
        ss += v * v;
    }
    float rs = 1.f / fmaxf(sqrtf(ss), 1e-12f);
    for (int c = 0; c < CIN; ++c) {
        float v = fabsf(w[(o * CIN + c) * 25 + tap]) * rs;
        Wb[((size_t)tap * COUT + o) * CIN + c] = (__bf16)v;
    }
}

// ---------------------------------------------------------------------------
// Kernel 2: s(b,h,w) = sum_c x^2  (channel-reduced square map)
// ---------------------------------------------------------------------------
__global__ void k_smap(const float* __restrict__ x, float* __restrict__ s) {
    int h  = blockIdx.x % H_;
    int b  = blockIdx.x / H_;
    int wv = threadIdx.x;                 // 0..127
    const float* xp = x + (((size_t)b * CIN) * H_ + h) * W_ + wv;
    float acc = 0.f;
    for (int c = 0; c < CIN; ++c) {
        float v = xp[(size_t)c * H_ * W_];
        acc += v * v;
    }
    s[((size_t)b * H_ + h) * W_ + wv] = acc;
}

// ---------------------------------------------------------------------------
// Kernel 3: NCHW fp32 -> NHWC bf16 (channel-contiguous) via LDS transpose
// ---------------------------------------------------------------------------
__global__ void k_xt(const float* __restrict__ x, __bf16* __restrict__ xT) {
    __shared__ __align__(16) __bf16 T[W_][72];
    int h   = blockIdx.x % H_;
    int b   = blockIdx.x / H_;
    int tid = threadIdx.x;                        // 0..255
    const float* xp = x + (((size_t)b * CIN) * H_ + h) * W_;
    for (int i = 0; i < (CIN * W_) / 256; ++i) {  // 32 iterations
        int lin = i * 256 + tid;
        int c   = lin >> 7;
        int wv  = lin & (W_ - 1);
        T[wv][c] = (__bf16)xp[(size_t)c * H_ * W_ + wv];
    }
    __syncthreads();
    int wv   = tid >> 1;
    int half = tid & 1;
    __bf16* dst = xT + (((size_t)b * H_ + h) * W_ + wv) * CIN + half * 32;
    const __bf16* srcp = &T[wv][half * 32];
    *(v8bf*)(dst +  0) = *(const v8bf*)(srcp +  0);
    *(v8bf*)(dst +  8) = *(const v8bf*)(srcp +  8);
    *(v8bf*)(dst + 16) = *(const v8bf*)(srcp + 16);
    *(v8bf*)(dst + 24) = *(const v8bf*)(srcp + 24);
}

// ---------------------------------------------------------------------------
// Kernel 4: implicit-GEMM conv via v_wmma_f32_16x16x32_bf16.
// Block = 256 threads (8 waves) = one full image row: 128 pixels x 64 cout.
// Wave grid 4(M) x 2(N); each wave: 2x2 register tile (32 pix x 32 cout,
// 4 accumulators) -> 2 loads per WMMA, 200 WMMAs per wave.
// Halo (5 x 132 pixel rows) staged once via async b128 copies; the halo LDS
// pool is reused as the fp32 output-transpose buffer after a barrier.
// ---------------------------------------------------------------------------
__global__ void __launch_bounds__(256, 2) k_main(
    const __bf16* __restrict__ xT, const __bf16* __restrict__ Wb,
    const float* __restrict__ s, float* __restrict__ y)
{
    __shared__ __align__(16) char pool[5 * HCOL * ASTR * 2];   // 95,040 B
    __shared__ float xsq[W_];
    __bf16 (*As)[HCOL][ASTR] = reinterpret_cast<__bf16 (*)[HCOL][ASTR]>(pool);
    float  (*yb)[HCOL]       = reinterpret_cast<float  (*)[HCOL]>(pool); // reused

    int tid  = threadIdx.x;
    int lane = tid & 31;
    int wave = tid >> 5;
    int wm   = wave & 3;        // M group: pixels [wm*32, wm*32+32)
    int wn   = wave >> 2;       // N group: cout  [wn*32, wn*32+32)

    int h = blockIdx.x & (H_ - 1);
    int b = blockIdx.x >> 7;

    // ---- stage halo: 5 x 132 pixel rows x 8 x 16B segments = 5280 tiles
    for (int it = 0; it < 21; ++it) {
        int lin = it * 256 + tid;
        if (lin < 5 * HCOL * 8) {
            int rowi = lin >> 3, seg = lin & 7;
            int kh   = rowi / HCOL;
            int col  = rowi - kh * HCOL;
            int rh = refl(h + kh - PADR, H_);
            int rw = refl(col - PADR, W_);
            const __bf16* src = xT + (((size_t)b * H_ + rh) * W_ + rw) * CIN + seg * 8;
            __bf16* dst = &As[kh][col][seg * 8];
#ifdef HAVE_ASYNC_LDS
            async_copy16(src, dst);
#else
            *(v8bf*)dst = *(const v8bf*)src;
#endif
        }
    }
#ifdef HAVE_ASYNC_LDS
    __builtin_amdgcn_s_wait_asynccnt(0);
#endif

    // ---- per-pixel patch sum-of-squares: 5x5 reflect window over s
    if (tid < W_) {
        float acc = 0.f;
        for (int dh = -PADR; dh <= PADR; ++dh) {
            const float* sp = s + ((size_t)b * H_ + refl(h + dh, H_)) * W_;
            for (int dw = -PADR; dw <= PADR; ++dw)
                acc += sp[refl(tid + dw, W_)];
        }
        xsq[tid] = acc;
    }
    __syncthreads();

    // ---- 25 taps x 2 cin-chunks; 2x2 WMMA register tile per wave
    v8f acc00 = {}, acc01 = {}, acc10 = {}, acc11 = {};
    int ml = lane & 15;
    int kb = (lane >> 4) << 3;                      // 0 or 8
    const __bf16* a0 = &As[0][wm * 32 + ml][kb];
    const __bf16* b0 = Wb + ((size_t)(wn * 32 + ml)) * CIN + kb;

    #pragma unroll
    for (int kh = 0; kh < 5; ++kh) {
        #pragma unroll
        for (int kw = 0; kw < 5; ++kw) {
            const __bf16* ar = a0 + (kh * HCOL + kw) * ASTR;
            const __bf16* br = b0 + (size_t)(kh * 5 + kw) * COUT * CIN;
            #pragma unroll
            for (int c0 = 0; c0 < CIN; c0 += 32) {
                v16bf aF0 = ld16(ar + c0);
                v16bf aF1 = ld16(ar + 16 * ASTR + c0);
                v16bf bF0 = ld16(br + c0);
                v16bf bF1 = ld16(br + 16 * CIN + c0);
                acc00 = __builtin_amdgcn_wmma_f32_16x16x32_bf16(
                    false, aF0, false, bF0, (short)0, acc00, false, false);
                acc01 = __builtin_amdgcn_wmma_f32_16x16x32_bf16(
                    false, aF0, false, bF1, (short)0, acc01, false, false);
                acc10 = __builtin_amdgcn_wmma_f32_16x16x32_bf16(
                    false, aF1, false, bF0, (short)0, acc10, false, false);
                acc11 = __builtin_amdgcn_wmma_f32_16x16x32_bf16(
                    false, aF1, false, bF1, (short)0, acc11, false, false);
            }
        }
    }
    __syncthreads();   // all halo reads done -> pool can be reused as yb

    // ---- cosine normalization + LDS transpose for coalesced NCHW store
    int hihalf = (lane >> 4) << 3;
    #pragma unroll
    for (int v = 0; v < 8; ++v) {
        int pmA = wm * 32 + v + hihalf;            // C layout: VGPR v -> pixel
        int pmB = pmA + 16;
        float r0 = rsqrtf(xsq[pmA] + 1e-8f);
        float r1 = rsqrtf(xsq[pmB] + 1e-8f);
        yb[wn * 32 + ml][pmA]      = acc00[v] * r0;
        yb[wn * 32 + 16 + ml][pmA] = acc01[v] * r0;
        yb[wn * 32 + ml][pmB]      = acc10[v] * r1;
        yb[wn * 32 + 16 + ml][pmB] = acc11[v] * r1;
    }
    __syncthreads();

    int co = tid >> 2, sg = tid & 3;               // 64 cout x 4 segments
    float* dsty = y + (((size_t)(b * COUT + co)) * H_ + h) * W_ + sg * 32;
    const float* srcy = &yb[co][sg * 32];
    #pragma unroll
    for (int q = 0; q < 8; ++q)
        *(float4*)(dsty + q * 4) = *(const float4*)(srcy + q * 4);
}

// ---------------------------------------------------------------------------
// Kernel 5: depthwise 5x5 DoG conv, zero padding
// ---------------------------------------------------------------------------
__global__ void k_dog(const float* __restrict__ y, float* __restrict__ out, SMK smk) {
    int h  = blockIdx.x & (H_ - 1);
    int oc = (blockIdx.x >> 7) & (COUT - 1);
    int b  = blockIdx.x >> 13;
    int wv = threadIdx.x;
    const float* yp = y + (size_t)(b * COUT + oc) * H_ * W_;
    float acc = 0.f;
    #pragma unroll
    for (int dh = -PADR; dh <= PADR; ++dh) {
        int hh = h + dh;
        if (hh < 0 || hh >= H_) continue;
        const float* row = yp + (size_t)hh * W_;
        #pragma unroll
        for (int dw = -PADR; dw <= PADR; ++dw) {
            int ww = wv + dw;
            if (ww < 0 || ww >= W_) continue;
            acc += smk.v[(dh + 2) * 5 + (dw + 2)] * row[ww];
        }
    }
    out[((size_t)(b * COUT + oc) * H_ + h) * W_ + wv] = acc;
}

// ---------------------------------------------------------------------------
extern "C" void kernel_launch(void* const* d_in, const int* in_sizes, int n_in,
                              void* d_out, int out_size, void* d_ws, size_t ws_size,
                              hipStream_t stream)
{
    const float* x   = (const float*)d_in[0];
    const float* wgt = (const float*)d_in[1];
    float* out = (float*)d_out;

    // workspace layout
    char* ws = (char*)d_ws;
    __bf16* Wb = (__bf16*)(ws);                                   // 200 KB
    float*  s  = (float*)(ws + 204800);                           // 1 MB
    __bf16* xT = (__bf16*)(ws + 204800 + 1048576);                // 32 MB
    float*  y  = (float*)(ws + 204800 + 1048576 + 33554432);      // 64 MB

    // DoG coefficients (host-side, deterministic)
    SMK smk;
    {
        const double PI = 3.14159265358979323846;
        const double se = 1.2, si = 1.4;
        const double ae = 1.0 / (2.0 * PI * se * se);
        const double ai = 1.0 / (2.0 * PI * si * si);
        const double ctr = ae - ai;
        for (int i = 0; i < 5; ++i)
            for (int j = 0; j < 5; ++j) {
                double dx = i - 2, dy = j - 2, r2 = dx * dx + dy * dy;
                double d = ae * std::exp(-r2 / (2.0 * se * se))
                         - ai * std::exp(-r2 / (2.0 * si * si));
                smk.v[i * 5 + j] = (float)(d / ctr);
            }
    }

    hipLaunchKernelGGL(k_wnorm, dim3(25),             dim3(64),  0, stream, wgt, Wb);
    hipLaunchKernelGGL(k_smap,  dim3(B_ * H_),        dim3(W_),  0, stream, x, s);
    hipLaunchKernelGGL(k_xt,    dim3(B_ * H_),        dim3(256), 0, stream, x, xT);
    hipLaunchKernelGGL(k_main,  dim3(B_ * H_),        dim3(256), 0, stream, xT, Wb, s, y);
    hipLaunchKernelGGL(k_dog,   dim3(B_ * COUT * H_), dim3(W_),  0, stream, y, out, smk);
}
